// PhysicsGAT_37434934952051
// MI455X (gfx1250) — compile-verified, matrix operand order checked by reading, not run on hardware
//
#include <hip/hip_runtime.h>
#include <hip/hip_bf16.h>

// ---------------- problem constants (match reference) ----------------
#define N_NODES 50000
#define N_EDGES 250000
#define EE      (N_EDGES + N_NODES)   // edges + self loops = 300000
#define B_GRAPHS 64
#define IN_F    5
#define H1      4
#define C1      64
#define D1      (H1 * C1)             // 256
#define C2      64
#define NEG_SLOPE 0.2f
#define ENC_NEG_INF 0x007FFFFFu       // enc_f32(-inf)

typedef __attribute__((ext_vector_type(16))) __bf16 bf16x16;
typedef __attribute__((ext_vector_type(8)))  float  v8f;

// order-preserving float <-> uint encoding for atomicMax-based segment max
__device__ __forceinline__ unsigned enc_f32(float f) {
  unsigned u = __float_as_uint(f);
  return (u & 0x80000000u) ? ~u : (u | 0x80000000u);
}
__device__ __forceinline__ float dec_f32(unsigned u) {
  return __uint_as_float((u & 0x80000000u) ? (u & 0x7fffffffu) : ~u);
}

// ---------------- generic fill ----------------
__global__ void k_fill_u32(unsigned* __restrict__ p, unsigned v, int n) {
  int i = blockIdx.x * blockDim.x + threadIdx.x;
  if (i < n) p[i] = v;
}

// ---------------- conv1 linear: xl1/xr1 = x @ W + b (K=5, memory bound) ----
__global__ void k_lin1(const float* __restrict__ x,
                       const float* __restrict__ Wl, const float* __restrict__ bl,
                       const float* __restrict__ Wr, const float* __restrict__ br,
                       float* __restrict__ xl, float* __restrict__ xr) {
  int idx = blockIdx.x * blockDim.x + threadIdx.x;   // node*256 + col
  if (idx >= N_NODES * D1) return;
  int n = idx >> 8, c = idx & 255;
  float xv[IN_F];
#pragma unroll
  for (int k = 0; k < IN_F; ++k) xv[k] = x[n * IN_F + k];
  float al = bl[c], ar = br[c];
#pragma unroll
  for (int k = 0; k < IN_F; ++k) {
    al = fmaf(xv[k], Wl[k * D1 + c], al);
    ar = fmaf(xv[k], Wr[k * D1 + c], ar);
  }
  xl[idx] = al;
  xr[idx] = ar;
}

// -------- edge logits: e = att . leaky_relu(xl[src]+xr[dst]); seg-max ------
__global__ void k_edge_logits(const float* __restrict__ xl, const float* __restrict__ xr,
                              const float* __restrict__ att, const int* __restrict__ ei,
                              float* __restrict__ e_out, unsigned* __restrict__ m_enc,
                              int H, int C) {
  int idx = blockIdx.x * blockDim.x + threadIdx.x;   // edge*H + h
  if (idx >= EE * H) return;
  int h = idx % H, edge = idx / H;
  int src, dst;
  if (edge < N_EDGES) { src = ei[edge]; dst = ei[N_EDGES + edge]; }
  else                { src = dst = edge - N_EDGES; }           // self loop
  const float* pl = xl + (size_t)src * H * C + (size_t)h * C;
  const float* pr = xr + (size_t)dst * H * C + (size_t)h * C;
  const float* pa = att + (size_t)h * C;
  float acc = 0.f;
  for (int c = 0; c < C; c += 4) {
    float4 l = *(const float4*)(pl + c);
    float4 r = *(const float4*)(pr + c);
    float4 a = *(const float4*)(pa + c);
    float z;
    z = l.x + r.x; z = z > 0.f ? z : NEG_SLOPE * z; acc = fmaf(z, a.x, acc);
    z = l.y + r.y; z = z > 0.f ? z : NEG_SLOPE * z; acc = fmaf(z, a.y, acc);
    z = l.z + r.z; z = z > 0.f ? z : NEG_SLOPE * z; acc = fmaf(z, a.z, acc);
    z = l.w + r.w; z = z > 0.f ? z : NEG_SLOPE * z; acc = fmaf(z, a.w, acc);
  }
  e_out[idx] = acc;
  atomicMax(&m_enc[(size_t)dst * H + h], enc_f32(acc));
}

// -------- softmax numerator + segment sum: w = exp(e - m[dst]) -------------
__global__ void k_edge_w(float* __restrict__ ew, const unsigned* __restrict__ m_enc,
                         float* __restrict__ s, const int* __restrict__ ei, int H) {
  int idx = blockIdx.x * blockDim.x + threadIdx.x;
  if (idx >= EE * H) return;
  int h = idx % H, edge = idx / H;
  int dst = (edge < N_EDGES) ? ei[N_EDGES + edge] : edge - N_EDGES;
  float m = dec_f32(m_enc[(size_t)dst * H + h]);
  float w = __expf(ew[idx] - m);
  ew[idx] = w;                                   // overwrite e with w
  atomicAdd(&s[(size_t)dst * H + h], w);
}

// -------- aggregation: out[dst] += (w/s[dst]) * xl[src] --------------------
__global__ void k_aggregate(const float* __restrict__ ew, const float* __restrict__ s,
                            const float* __restrict__ xl, const int* __restrict__ ei,
                            float* __restrict__ out, int H, int C) {
  int c4n = C >> 2;
  int idx = blockIdx.x * blockDim.x + threadIdx.x;  // (edge*H + h)*c4n + c4
  if (idx >= EE * H * c4n) return;
  int c4 = idx % c4n, t = idx / c4n;
  int h = t % H, edge = t / H;
  int src, dst;
  if (edge < N_EDGES) { src = ei[edge]; dst = ei[N_EDGES + edge]; }
  else                { src = dst = edge - N_EDGES; }
  float alpha = ew[t] / s[(size_t)dst * H + h];
  float4 xv = *(const float4*)(xl + (size_t)src * H * C + (size_t)h * C + c4 * 4);
  float* po = out + (size_t)dst * H * C + (size_t)h * C + c4 * 4;
  atomicAdd(po + 0, alpha * xv.x);
  atomicAdd(po + 1, alpha * xv.y);
  atomicAdd(po + 2, alpha * xv.z);
  atomicAdd(po + 3, alpha * xv.w);
}

// -------- bias + ELU -> bf16 (feeds the WMMA GEMM) -------------------------
__global__ void k_bias_elu_bf16(const float* __restrict__ in, const float* __restrict__ bias,
                                __hip_bfloat16* __restrict__ ob, int D, int total) {
  int i = blockIdx.x * blockDim.x + threadIdx.x;
  if (i >= total) return;
  float v = in[i] + bias[i % D];
  v = v > 0.f ? v : (__expf(v) - 1.f);
  ob[i] = __float2bfloat16(v);
}

// -------- bias + ELU in-place f32 ------------------------------------------
__global__ void k_bias_elu_f32(float* __restrict__ io, const float* __restrict__ bias,
                               int D, int total) {
  int i = blockIdx.x * blockDim.x + threadIdx.x;
  if (i >= total) return;
  float v = io[i] + bias[i % D];
  io[i] = v > 0.f ? v : (__expf(v) - 1.f);
}

// -------- transpose W [256,64] f32 -> Wt [64,256] bf16 ---------------------
__global__ void k_wt(const float* __restrict__ W, __hip_bfloat16* __restrict__ Wt) {
  int i = blockIdx.x * blockDim.x + threadIdx.x;   // k*64 + n
  if (i >= D1 * C2) return;
  int k = i / C2, n = i % C2;
  Wt[(size_t)n * D1 + k] = __float2bfloat16(W[i]);
}

// -------- fused WMMA GEMM: {CL,CR}[N,64] = A[N,256](bf16) @ {WtL,WtR}^T + b -
// One wave owns a 16-row strip and produces the full 16x64 output tile for
// BOTH weight matrices: the A fragment is loaded once per k-step and reused
// by 8 WMMAs (4 column tiles x 2 matrices).  B totals 64 KB bf16 -> L0/L2
// resident, so its reloads are cheap; A streams through exactly once.
//
// A/B fragments follow the CDNA5 16-bit 16x32 layout:
//   lanes 0-15  : row/col = lane,    K in {k0+0..7,  k0+16..23}
//   lanes 16-31 : row/col = lane-16, K in {k0+8..15, k0+24..31}
// Both A rows and pre-transposed B columns are contiguous in K, so each
// fragment is two 16-byte global loads.
__global__ void __launch_bounds__(32) k_gemm_bf16_fused(
    const __hip_bfloat16* __restrict__ A,     // [N_NODES, 256] row major
    const __hip_bfloat16* __restrict__ BtL,   // [64, 256] (col n, contiguous K)
    const __hip_bfloat16* __restrict__ BtR,   // [64, 256]
    const float* __restrict__ biasL,          // [64]
    const float* __restrict__ biasR,          // [64]
    float* __restrict__ CL,                   // [N_NODES, 64]
    float* __restrict__ CR) {                 // [N_NODES, 64]
  const int lane = threadIdx.x & 31;
  const int half = lane >> 4;
  const int l15  = lane & 15;
  const int koff = half * 8;
  const int m0   = blockIdx.x * 16;

  const __hip_bfloat16* ap = A + (size_t)(m0 + l15) * D1;

  union Frag { bf16x16 v; uint4 q[2]; };
  Frag a;
  v8f acc[8];                                  // [0..3]=L tiles, [4..7]=R tiles
#pragma unroll
  for (int nt = 0; nt < 4; ++nt) {
    float bl = biasL[nt * 16 + l15];
    float br = biasR[nt * 16 + l15];
#pragma unroll
    for (int r = 0; r < 8; ++r) { acc[nt][r] = bl; acc[4 + nt][r] = br; }
  }

#pragma unroll
  for (int k0 = 0; k0 < D1; k0 += 32) {
    a.q[0] = *reinterpret_cast<const uint4*>(ap + k0 + koff);
    a.q[1] = *reinterpret_cast<const uint4*>(ap + k0 + 16 + koff);
#pragma unroll
    for (int nt = 0; nt < 4; ++nt) {
      Frag b;
      const __hip_bfloat16* bpL = BtL + (size_t)(nt * 16 + l15) * D1;
      b.q[0] = *reinterpret_cast<const uint4*>(bpL + k0 + koff);
      b.q[1] = *reinterpret_cast<const uint4*>(bpL + k0 + 16 + koff);
      acc[nt] = __builtin_amdgcn_wmma_f32_16x16x32_bf16(
          false, a.v, false, b.v, (short)0, acc[nt], false, false);
      const __hip_bfloat16* bpR = BtR + (size_t)(nt * 16 + l15) * D1;
      b.q[0] = *reinterpret_cast<const uint4*>(bpR + k0 + koff);
      b.q[1] = *reinterpret_cast<const uint4*>(bpR + k0 + 16 + koff);
      acc[4 + nt] = __builtin_amdgcn_wmma_f32_16x16x32_bf16(
          false, a.v, false, b.v, (short)0, acc[4 + nt], false, false);
    }
  }

  // C/D layout: VGPR r, lane -> row = m0 + r + 8*half, col = nt*16 + (lane&15)
#pragma unroll
  for (int nt = 0; nt < 4; ++nt) {
#pragma unroll
    for (int r = 0; r < 8; ++r) {
      size_t o = (size_t)(m0 + r + 8 * half) * C2 + nt * 16 + l15;
      CL[o] = acc[nt][r];
      CR[o] = acc[4 + nt][r];
    }
  }
}

// -------- global mean/max/add pooling via atomics --------------------------
__global__ void k_pool(const float* __restrict__ h, const int* __restrict__ batch,
                       float* __restrict__ psum, unsigned* __restrict__ pmax,
                       int* __restrict__ pcnt) {
  int idx = blockIdx.x * blockDim.x + threadIdx.x;  // node*64 + c
  if (idx >= N_NODES * C2) return;
  int n = idx >> 6, c = idx & 63;
  int b = batch[n];
  float v = h[idx];
  atomicAdd(&psum[b * C2 + c], v);
  atomicMax(&pmax[b * C2 + c], enc_f32(v));
  if (c == 0) atomicAdd(&pcnt[b], 1);
}

// -------- final linear: [mean|max|sum] @ Wlin + blin -----------------------
__global__ void k_final(const float* __restrict__ psum, const unsigned* __restrict__ pmax,
                        const int* __restrict__ pcnt, const float* __restrict__ Wlin,
                        const float* __restrict__ blin, float* __restrict__ out) {
  int idx = blockIdx.x * blockDim.x + threadIdx.x;  // b*2 + o
  if (idx >= B_GRAPHS * 2) return;
  int b = idx >> 1, o = idx & 1;
  float cnt = (float)(pcnt[b] > 1 ? pcnt[b] : 1);
  float acc = blin[o];
  for (int c = 0; c < C2; ++c) {
    float sm = psum[b * C2 + c];
    float mean = sm / cnt;
    float mx = dec_f32(pmax[b * C2 + c]);
    acc = fmaf(mean, Wlin[c * 2 + o], acc);
    acc = fmaf(mx,   Wlin[(C2 + c) * 2 + o], acc);
    acc = fmaf(sm,   Wlin[(2 * C2 + c) * 2 + o], acc);
  }
  out[idx] = acc;
}

// ===========================================================================
extern "C" void kernel_launch(void* const* d_in, const int* in_sizes, int n_in,
                              void* d_out, int out_size, void* d_ws, size_t ws_size,
                              hipStream_t stream) {
  (void)in_sizes; (void)n_in; (void)out_size; (void)ws_size;
  const float* x     = (const float*)d_in[0];
  const int*   ei    = (const int*)d_in[1];
  const int*   batch = (const int*)d_in[2];
  const float* Wl1   = (const float*)d_in[3];
  const float* bl1   = (const float*)d_in[4];
  const float* Wr1   = (const float*)d_in[5];
  const float* br1   = (const float*)d_in[6];
  const float* att1  = (const float*)d_in[7];
  const float* bias1 = (const float*)d_in[8];
  const float* Wl2   = (const float*)d_in[9];
  const float* bl2   = (const float*)d_in[10];
  const float* Wr2   = (const float*)d_in[11];
  const float* br2   = (const float*)d_in[12];
  const float* att2  = (const float*)d_in[13];
  const float* bias2 = (const float*)d_in[14];
  const float* Wlin  = (const float*)d_in[15];
  const float* blin  = (const float*)d_in[16];
  float* out = (float*)d_out;

  // ---- workspace layout ----
  char* ws = (char*)d_ws;
  size_t off = 0;
  auto alloc = [&](size_t bytes) -> char* {
    char* p = ws + off;
    off = (off + bytes + 255) & ~(size_t)255;
    return p;
  };
  float*          xl1  = (float*)alloc((size_t)N_NODES * D1 * 4);
  float*          xr1  = (float*)alloc((size_t)N_NODES * D1 * 4);
  float*          ew1  = (float*)alloc((size_t)EE * H1 * 4);
  unsigned*       m1   = (unsigned*)alloc((size_t)N_NODES * H1 * 4);
  float*          s1   = (float*)alloc((size_t)N_NODES * H1 * 4);
  float*          agg1 = (float*)alloc((size_t)N_NODES * D1 * 4);
  __hip_bfloat16* hb   = (__hip_bfloat16*)alloc((size_t)N_NODES * D1 * 2);
  __hip_bfloat16* wtl  = (__hip_bfloat16*)alloc((size_t)C2 * D1 * 2);
  __hip_bfloat16* wtr  = (__hip_bfloat16*)alloc((size_t)C2 * D1 * 2);
  float*          xl2  = (float*)alloc((size_t)N_NODES * C2 * 4);
  float*          xr2  = (float*)alloc((size_t)N_NODES * C2 * 4);
  float*          ew2  = (float*)alloc((size_t)EE * 4);
  unsigned*       m2   = (unsigned*)alloc((size_t)N_NODES * 4);
  float*          s2   = (float*)alloc((size_t)N_NODES * 4);
  float*          agg2 = (float*)alloc((size_t)N_NODES * C2 * 4);
  float*          psum = (float*)alloc((size_t)B_GRAPHS * C2 * 4);
  unsigned*       pmax = (unsigned*)alloc((size_t)B_GRAPHS * C2 * 4);
  int*            pcnt = (int*)alloc((size_t)B_GRAPHS * 4);

  const int T = 256;
  auto G = [](long n, int t) { return (unsigned)((n + t - 1) / t); };

  // ---- re-init accumulators (graph-replay safe) ----
  k_fill_u32<<<G(N_NODES * H1, T), T, 0, stream>>>(m1, ENC_NEG_INF, N_NODES * H1);
  k_fill_u32<<<G(N_NODES * H1, T), T, 0, stream>>>((unsigned*)s1, 0u, N_NODES * H1);
  k_fill_u32<<<G((long)N_NODES * D1, T), T, 0, stream>>>((unsigned*)agg1, 0u, N_NODES * D1);
  k_fill_u32<<<G(N_NODES, T), T, 0, stream>>>(m2, ENC_NEG_INF, N_NODES);
  k_fill_u32<<<G(N_NODES, T), T, 0, stream>>>((unsigned*)s2, 0u, N_NODES);
  k_fill_u32<<<G((long)N_NODES * C2, T), T, 0, stream>>>((unsigned*)agg2, 0u, N_NODES * C2);
  k_fill_u32<<<G(B_GRAPHS * C2, T), T, 0, stream>>>((unsigned*)psum, 0u, B_GRAPHS * C2);
  k_fill_u32<<<G(B_GRAPHS * C2, T), T, 0, stream>>>(pmax, ENC_NEG_INF, B_GRAPHS * C2);
  k_fill_u32<<<G(B_GRAPHS, T), T, 0, stream>>>((unsigned*)pcnt, 0u, B_GRAPHS);

  // ---- layer 1 ----
  k_lin1<<<G((long)N_NODES * D1, T), T, 0, stream>>>(x, Wl1, bl1, Wr1, br1, xl1, xr1);
  k_edge_logits<<<G((long)EE * H1, T), T, 0, stream>>>(xl1, xr1, att1, ei, ew1, m1, H1, C1);
  k_edge_w<<<G((long)EE * H1, T), T, 0, stream>>>(ew1, m1, s1, ei, H1);
  k_aggregate<<<G((long)EE * H1 * (C1 / 4), T), T, 0, stream>>>(ew1, s1, xl1, ei, agg1, H1, C1);
  k_bias_elu_bf16<<<G((long)N_NODES * D1, T), T, 0, stream>>>(agg1, bias1, hb, D1, N_NODES * D1);

  // ---- layer 2 dense transforms via fused WMMA GEMM ----
  k_wt<<<G(D1 * C2, T), T, 0, stream>>>(Wl2, wtl);
  k_wt<<<G(D1 * C2, T), T, 0, stream>>>(Wr2, wtr);
  k_gemm_bf16_fused<<<N_NODES / 16, 32, 0, stream>>>(hb, wtl, wtr, bl2, br2, xl2, xr2);

  // ---- layer 2 attention + aggregation ----
  k_edge_logits<<<G((long)EE, T), T, 0, stream>>>(xl2, xr2, att2, ei, ew2, m2, 1, C2);
  k_edge_w<<<G((long)EE, T), T, 0, stream>>>(ew2, m2, s2, ei, 1);
  k_aggregate<<<G((long)EE * (C2 / 4), T), T, 0, stream>>>(ew2, s2, xl2, ei, agg2, 1, C2);
  k_bias_elu_f32<<<G((long)N_NODES * C2, T), T, 0, stream>>>(agg2, bias2, C2, N_NODES * C2);

  // ---- pooling + head ----
  k_pool<<<G((long)N_NODES * C2, T), T, 0, stream>>>(agg2, batch, psum, pmax, pcnt);
  k_final<<<1, 128, 0, stream>>>(psum, pmax, pcnt, Wlin, blin, out);
}
